// GraphNetwork_76725295776241
// MI455X (gfx1250) — matrix operand order repaired and checked by reading, not compiled, not run on hardware
//
#include <hip/hip_runtime.h>
#include <hip/hip_bf16.h>
#include <math.h>

// ---------------------------------------------------------------------------
// GraphNetwork (2x GCNConv + edge-pair MLP) for MI455X / gfx1250 (wave32).
// Dominant cost: [E,256]@[256,128] edge MLP (52 GF) -> v_wmma_f32_16x16x32_f16.
// Node table (12.8MB f16) is L2-resident (192MB L2), so edge gathers are L2 hits.
// Edge MLP: 128 edges/block, lin1^T held in registers (B reuse x8 vs round 1).
// ---------------------------------------------------------------------------

typedef __attribute__((ext_vector_type(16))) _Float16 v16h;
typedef __attribute__((ext_vector_type(8)))  _Float16 v8h;
typedef __attribute__((ext_vector_type(8)))  float    v8f;

// A-matrix fragment (16-bit A 16x32): lane hi=lane>>4.
// elems 0..15 = K (hi*16 + 0..15), matching B's sequential-K layout.
// Caller passes p = row_base + ktile*32 + hi*16.
__device__ __forceinline__ v16h load_a_frag(const _Float16* p) {
    return *(const v16h*)(p);
}

__device__ __forceinline__ v8f wmma_f16(v16h a, v16h b, v8f c) {
    return __builtin_amdgcn_wmma_f32_16x16x32_f16(
        /*neg_a=*/false, a, /*neg_b=*/false, b,
        /*c_mod=*/(short)0, c, /*reuse_a=*/false, /*reuse_b=*/false);
}

// ------------------------------- prep kernels ------------------------------

__global__ void k_fill1(float* p, int n) {
    int i = blockIdx.x * blockDim.x + threadIdx.x;
    if (i < n) p[i] = 1.0f;                     // self-loop degree
}

__global__ void k_deg(const int* __restrict__ ei, float* deg, int E) {
    int e = blockIdx.x * blockDim.x + threadIdx.x;
    if (e < E) atomicAdd(&deg[ei[E + e]], 1.0f); // count over targets (col)
}

__global__ void k_rsqrt(float* p, int n) {
    int i = blockIdx.x * blockDim.x + threadIdx.x;
    if (i < n) p[i] = rsqrtf(p[i]);
}

// WT[o*K + k] = (f16) W[k*Dout + o]   (transpose + f32->f16)
__global__ void k_transpose_cvt(const float* __restrict__ W, _Float16* __restrict__ WT,
                                int K, int Dout) {
    int i = blockIdx.x * blockDim.x + threadIdx.x;
    if (i >= K * Dout) return;
    int o = i / K, k = i - o * K;
    WT[i] = (_Float16)W[(size_t)k * Dout + o];
}

// h = feat @ W1   (N x 8 @ 8 x 128) - K=8, scalar is fine
__global__ void k_gemm_in8(const float* __restrict__ feat, const float* __restrict__ W1,
                           float* __restrict__ h, int N) {
    int i = blockIdx.x * blockDim.x + threadIdx.x;
    if (i >= N * 128) return;
    int n = i >> 7, d = i & 127;
    float acc = 0.f;
#pragma unroll
    for (int k = 0; k < 8; ++k) acc += feat[n * 8 + k] * W1[k * 128 + d];
    h[i] = acc;
}

// acc[n][d] = h[n][d] * dinv[n]^2   (self-loop term of symmetric norm)
__global__ void k_self(const float* __restrict__ h, const float* __restrict__ dinv,
                       float* __restrict__ acc, int N) {
    int i = blockIdx.x * blockDim.x + threadIdx.x;
    if (i >= N * 128) return;
    float dv = dinv[i >> 7];
    acc[i] = h[i] * dv * dv;
}

// scatter: acc[col] += h[row] * dinv[row]*dinv[col]; one wave per edge, float4/lane
__global__ void k_scatter(const float* __restrict__ h, const float* __restrict__ dinv,
                          const int* __restrict__ ei, float* acc, int E) {
    int t = blockIdx.x * blockDim.x + threadIdx.x;
    int e = t >> 5;
    if (e >= E) return;
    int lane = t & 31;
    int row = ei[e], col = ei[E + e];
    float nrm = dinv[row] * dinv[col];
    float4 v = *(const float4*)(h + (size_t)row * 128 + lane * 4);
    float* dst = acc + (size_t)col * 128 + lane * 4;
    atomicAdd(dst + 0, v.x * nrm);
    atomicAdd(dst + 1, v.y * nrm);
    atomicAdd(dst + 2, v.z * nrm);
    atomicAdd(dst + 3, v.w * nrm);
}

// x16 = (f16) relu(acc + bias)
__global__ void k_relu_cvt(const float* __restrict__ acc, const float* __restrict__ bias,
                           _Float16* __restrict__ x16, int N) {
    int i = blockIdx.x * blockDim.x + threadIdx.x;
    if (i >= N * 128) return;
    x16[i] = (_Float16)fmaxf(acc[i] + bias[i & 127], 0.f);
}

// ------------------------- GCN2 GEMM (N x 128 @ 128 x 128) -----------------
// block = 128 threads (4 waves); wave w computes cols [w*32, w*32+32) of a
// 16-row node tile; K=128 -> 4 WMMA K-steps per column tile.
__global__ __launch_bounds__(128)
void k_gcn2_gemm(const _Float16* __restrict__ x16, const _Float16* __restrict__ W2T,
                 float* __restrict__ out, int N) {
    int wave = threadIdx.x >> 5, lane = threadIdx.x & 31;
    int m = lane & 15, hi = lane >> 4;
    int row = blockIdx.x * 16 + m;
    if (row >= N) row = N - 1;                  // clamp (EXEC must stay all-1 for WMMA)
    const _Float16* pa  = x16 + (size_t)row * 128;
    int col0 = wave * 32 + m, col1 = col0 + 16;
    const _Float16* pb0 = W2T + (size_t)col0 * 128;
    const _Float16* pb1 = W2T + (size_t)col1 * 128;
    v8f c0 = {}, c1 = {};
#pragma unroll
    for (int t = 0; t < 4; ++t) {
        int ka = t * 32 + hi * 16;              // A sequential-K layout
        int kb = t * 32 + hi * 16;              // B sequential-K layout
        v16h a  = load_a_frag(pa + ka);
        v16h b0 = *(const v16h*)(pb0 + kb);
        c0 = wmma_f16(a, b0, c0);
        v16h b1 = *(const v16h*)(pb1 + kb);
        c1 = wmma_f16(a, b1, c1);
    }
    int orow0 = blockIdx.x * 16 + 8 * hi;       // C/D: lane holds col, VGPR r -> row
#pragma unroll
    for (int r = 0; r < 8; ++r) {
        int orow = orow0 + r;
        if (orow < N) {
            out[(size_t)orow * 128 + col0] = c0[r];
            out[(size_t)orow * 128 + col1] = c1[r];
        }
    }
}

// -------------------- edge MLP: relu([x_s||x_d]@lin1)+head -----------------
// 128 edges/block (8 row tiles of 16). Each wave owns 2 column tiles of B and
// keeps all 8 K-step fragments in registers (128 VGPRs) across the row loop,
// so the steady state is 8 A-gather loads + 16 WMMAs per row tile, zero B
// traffic. z goes to a 128x130 LDS buffer (pad -> bank-conflict-free), then
// each thread owns one edge for the 128->2 head + log-softmax.
#define EPB 128
__global__ __launch_bounds__(128)
void k_edge_mlp(const _Float16* __restrict__ x16, const int* __restrict__ ei,
                const _Float16* __restrict__ lin1T, const float* __restrict__ lin1b,
                const float* __restrict__ linfW, const float* __restrict__ linfb,
                float* __restrict__ out, int E) {
    __shared__ float zbuf[EPB][130];
    int wave = threadIdx.x >> 5, lane = threadIdx.x & 31;
    int m = lane & 15, hi = lane >> 4;
    int col0 = wave * 32 + m, col1 = col0 + 16;

    // resident B fragments: 8 K-tiles x 2 column tiles
    const _Float16* pb0 = lin1T + (size_t)col0 * 256;
    const _Float16* pb1 = lin1T + (size_t)col1 * 256;
    v16h B0[8], B1[8];
#pragma unroll
    for (int t = 0; t < 8; ++t) {
        int kb = t * 32 + hi * 16;              // B sequential-K layout
        B0[t] = *(const v16h*)(pb0 + kb);
        B1[t] = *(const v16h*)(pb1 + kb);
    }
    float bb0 = lin1b[col0], bb1 = lin1b[col1];

#pragma unroll 2
    for (int r = 0; r < EPB / 16; ++r) {
        int e = blockIdx.x * EPB + r * 16 + m;
        if (e >= E) e = E - 1;                  // clamp (EXEC all-1 for WMMA)
        int src = ei[e], dst = ei[E + e];
        const _Float16* ps = x16 + (size_t)src * 128;
        const _Float16* pd = x16 + (size_t)dst * 128;
        v8f c0 = {}, c1 = {};
#pragma unroll
        for (int t = 0; t < 8; ++t) {           // K = 256: src half then dst half
            const _Float16* pa = (t < 4) ? ps : pd;
            v16h a = load_a_frag(pa + (t & 3) * 32 + hi * 16);
            c0 = wmma_f16(a, B0[t], c0);
            c1 = wmma_f16(a, B1[t], c1);
        }
#pragma unroll
        for (int q = 0; q < 8; ++q) {           // C/D: lane holds col, VGPR q -> row
            zbuf[r * 16 + q + 8 * hi][col0] = fmaxf(c0[q] + bb0, 0.f);
            zbuf[r * 16 + q + 8 * hi][col1] = fmaxf(c1[q] + bb1, 0.f);
        }
    }
    __syncthreads();

    // head: thread t owns edge t of the block; both classes + log-softmax local
    int el = threadIdx.x;
    int eo = blockIdx.x * EPB + el;
    float a0 = linfb[0], a1 = linfb[1];
    const float2* wf = (const float2*)linfW;    // [128] x (c0,c1)
#pragma unroll 4
    for (int k = 0; k < 128; ++k) {
        float z  = zbuf[el][k];
        float2 w = wf[k];
        a0 += z * w.x;
        a1 += z * w.y;
    }
    if (eo < E) {
        float mx  = fmaxf(a0, a1);
        float lse = mx + logf(expf(a0 - mx) + expf(a1 - mx));
        out[(size_t)eo * 2 + 0] = a0 - lse;
        out[(size_t)eo * 2 + 1] = a1 - lse;
    }
}

// ------------------------- bbox pairs + index pairs ------------------------
// index pairs emitted as exact float values (N < 2^24)
__global__ void k_tail(const float* __restrict__ bbox, const int* __restrict__ bidx,
                       const int* __restrict__ ei, float* __restrict__ out, int E) {
    int e = blockIdx.x * blockDim.x + threadIdx.x;
    if (e >= E) return;
    int s = ei[e], d = ei[E + e];
    float4 bs = *(const float4*)(bbox + (size_t)s * 4);
    float4 bd = *(const float4*)(bbox + (size_t)d * 4);
    float* ob = out + (size_t)E * 2 + (size_t)e * 8;
    ob[0] = bs.x; ob[1] = bs.y; ob[2] = bs.z; ob[3] = bs.w;
    ob[4] = bd.x; ob[5] = bd.y; ob[6] = bd.z; ob[7] = bd.w;
    float* oi = out + (size_t)E * 10 + (size_t)e * 2;
    oi[0] = (float)bidx[s];
    oi[1] = (float)bidx[d];
}

// --------------------------------- launcher --------------------------------

extern "C" void kernel_launch(void* const* d_in, const int* in_sizes, int n_in,
                              void* d_out, int out_size, void* d_ws, size_t ws_size,
                              hipStream_t stream) {
    const float* feat   = (const float*)d_in[0];
    const float* bboxes = (const float*)d_in[1];
    const int*   bidx   = (const int*)  d_in[2];
    const int*   ei     = (const int*)  d_in[3];
    const float* W1     = (const float*)d_in[4];
    const float* b1     = (const float*)d_in[5];
    const float* W2     = (const float*)d_in[6];
    const float* b2     = (const float*)d_in[7];
    const float* lin1W  = (const float*)d_in[8];
    const float* lin1b  = (const float*)d_in[9];
    const float* linfW  = (const float*)d_in[10];
    const float* linfb  = (const float*)d_in[11];
    const int N = in_sizes[2];          // bbox_indices count
    const int E = in_sizes[3] / 2;      // edge_index is [2,E]
    float* outp = (float*)d_out;

    // workspace carve-out (256B aligned): ~64.3 MB total
    char* ws = (char*)d_ws;
    size_t off = 0;
    auto carve = [&](size_t bytes) -> void* {
        void* p = (void*)(ws + off);
        off = (off + bytes + 255) & ~(size_t)255;
        return p;
    };
    float*    bufA  = (float*)   carve((size_t)N * 128 * 4); // dense GEMM out (h1, then h2)
    float*    bufB  = (float*)   carve((size_t)N * 128 * 4); // aggregation accumulator
    _Float16* x16   = (_Float16*)carve((size_t)N * 128 * 2); // node features f16
    float*    dinv  = (float*)   carve((size_t)N * 4);       // deg -> rsqrt(deg)
    _Float16* W2T   = (_Float16*)carve(128 * 128 * 2);
    _Float16* lin1T = (_Float16*)carve(256 * 128 * 2);

    const int T = 256;
    auto g = [&](long long n) { return (unsigned)((n + T - 1) / T); };

    // normalization: deg = 1 + in-degree; dinv = rsqrt(deg)
    k_fill1 <<<g(N), T, 0, stream>>>(dinv, N);
    k_deg   <<<g(E), T, 0, stream>>>(ei, dinv, E);
    k_rsqrt <<<g(N), T, 0, stream>>>(dinv, N);

    // weight transpose + f16 convert
    k_transpose_cvt<<<g(128 * 128), T, 0, stream>>>(W2,    W2T,   128, 128);
    k_transpose_cvt<<<g(256 * 128), T, 0, stream>>>(lin1W, lin1T, 256, 128);

    // GCN layer 1
    k_gemm_in8 <<<g((long long)N * 128), T, 0, stream>>>(feat, W1, bufA, N);
    k_self     <<<g((long long)N * 128), T, 0, stream>>>(bufA, dinv, bufB, N);
    k_scatter  <<<g((long long)E * 32),  T, 0, stream>>>(bufA, dinv, ei, bufB, E);
    k_relu_cvt <<<g((long long)N * 128), T, 0, stream>>>(bufB, b1, x16, N);

    // GCN layer 2 (WMMA GEMM)
    k_gcn2_gemm<<<(N + 15) / 16, 128, 0, stream>>>(x16, W2T, bufA, N);
    k_self     <<<g((long long)N * 128), T, 0, stream>>>(bufA, dinv, bufB, N);
    k_scatter  <<<g((long long)E * 32),  T, 0, stream>>>(bufA, dinv, ei, bufB, E);
    k_relu_cvt <<<g((long long)N * 128), T, 0, stream>>>(bufB, b2, x16, N);

    // edge-pair MLP (dominant WMMA GEMM) + head + log-softmax
    k_edge_mlp<<<(E + EPB - 1) / EPB, 128, 0, stream>>>(x16, ei, lin1T, lin1b,
                                                        linfW, linfb, outp, E);
    // bbox pairs + index pairs
    k_tail<<<g(E), T, 0, stream>>>(bboxes, bidx, ei, outp, E);
}
